// WLGNN_86371792322833
// MI455X (gfx1250) — matrix operand order repaired and checked by reading, not compiled, not run on hardware
//
#include <hip/hip_runtime.h>
#include <hip/hip_bf16.h>

typedef __attribute__((ext_vector_type(16))) _Float16 v16h;
typedef __attribute__((ext_vector_type(8)))  float    v8f;

#define DF 64
#define TB 256

// ---------- small utility kernels ----------

__global__ void k_deg_init(float* __restrict__ deg, int n) {
  int i = blockIdx.x * blockDim.x + threadIdx.x;
  if (i < n) deg[i] = 1.0f;   // self-loop
}

__global__ void k_deg_count(const int* __restrict__ dst, float* __restrict__ deg, int e) {
  int i = blockIdx.x * blockDim.x + threadIdx.x;
  if (i < e) atomicAdd(&deg[dst[i]], 1.0f);
}

// Pack W[K,64] (f32) into WMMA B-fragment layout, f16:
// Bp[((k32*4 + wave)*32 + lane)*16 + j] = W[k32*32 + (lane>>4)*16 + j][wave*16 + (lane&15)]
__global__ void k_pack_B(const float* __restrict__ W, _Float16* __restrict__ Bp, int K) {
  int t = blockIdx.x * blockDim.x + threadIdx.x;
  int total = (K / 32) * 2048;
  if (t >= total) return;
  int j    = t & 15;
  int lane = (t >> 4) & 31;
  int wave = (t >> 9) & 3;
  int k32  = t >> 11;
  int lo = lane & 15, hi = lane >> 4;
  int col  = wave * 16 + lo;
  int krow = k32 * 32 + hi * 16 + j;
  Bp[t] = (_Float16)W[krow * DF + col];
}

// ---------- GraphNorm: stats over embedding gather ----------
__global__ void k_stats_emb(const int* __restrict__ x, const float* __restrict__ emb,
                            float* __restrict__ sum, float* __restrict__ sumsq, int n) {
  __shared__ float s1[TB], s2[TB];
  int t = threadIdx.x;
  int idx = blockIdx.x * TB + t;
  float v = 0.f;
  if (idx < n) {
    int row = idx >> 6, col = idx & 63;
    v = emb[x[row] * DF + col];
  }
  s1[t] = v; s2[t] = v * v;
  __syncthreads();
  if (t < 128) { s1[t] += s1[t + 128]; s2[t] += s2[t + 128]; }
  __syncthreads();
  if (t < 64) {
    atomicAdd(&sum[t],   s1[t] + s1[t + 64]);
    atomicAdd(&sumsq[t], s2[t] + s2[t + 64]);
  }
}

// mul/add precompute: out = mul[c]*x + add[c]
__global__ void k_finalize(const float* __restrict__ sum, const float* __restrict__ sumsq,
                           const float* __restrict__ alpha, const float* __restrict__ gamma,
                           const float* __restrict__ beta,
                           float* __restrict__ mul, float* __restrict__ add, float inv_n) {
  int c = threadIdx.x;
  if (c < DF) {
    float m   = sum[c] * inv_n;
    float ex2 = sumsq[c] * inv_n;
    float a   = alpha[c];
    float v   = ex2 - (2.0f * a - a * a) * m * m;   // E[(x-a m)^2]
    float sc  = gamma[c] * rsqrtf(v + 1e-5f);
    mul[c] = sc;
    add[c] = beta[c] - sc * a * m;
  }
}

__global__ void k_norm0(const int* __restrict__ x, const float* __restrict__ emb,
                        const float* __restrict__ mul, const float* __restrict__ add,
                        _Float16* __restrict__ out, int n) {
  int idx = blockIdx.x * blockDim.x + threadIdx.x;
  if (idx < n) {
    int row = idx >> 6, col = idx & 63;
    float v = emb[x[row] * DF + col];
    out[idx] = (_Float16)(mul[col] * v + add[col]);
  }
}

// ---------- WMMA GEMM: C[M,64] = A[M,K] (f16) @ W (pre-packed f16), f32 accumulate ----------
// 4 waves / block: wave owns a 16-col slab; block owns 4 M-tiles (64 rows).
// B fragments loaded once into registers, reused across all M tiles; K fully unrolled.
template <int K>
__global__ __launch_bounds__(128) void k_gemm_wmma(const _Float16* __restrict__ A,
                                                   const _Float16* __restrict__ Bp,
                                                   float* __restrict__ C, int M) {
  constexpr int KC = K / 32;
  int lane = threadIdx.x & 31;
  int wave = threadIdx.x >> 5;
  int hi = lane >> 4;       // half-wave select
  int lo = lane & 15;
  int n0 = wave * 16;

  union BU { v16h v; uint4 q[2]; };
  union AU { v16h v; uint4 q[2]; };

  // per-chunk B fragment: 2 contiguous 16B loads per lane
  BU b[KC];
  const uint4* bq = (const uint4*)Bp + ((size_t)wave * 32 + lane) * 2;
#pragma unroll
  for (int k = 0; k < KC; ++k) {
    b[k].q[0] = bq[(size_t)k * 256];
    b[k].q[1] = bq[(size_t)k * 256 + 1];
  }

  int mbase = blockIdx.x * 64;
#pragma unroll
  for (int mt = 0; mt < 4; ++mt) {
    int m0 = mbase + mt * 16;
    if (m0 < M) {
      const _Float16* Arow = A + (size_t)(m0 + lo) * K;
      v8f c = {};
#pragma unroll
      for (int k = 0; k < KC; ++k) {
        AU a;
        // A fragment in this code's k order: 16 contiguous k per lane (k = 16*hi + s), matching packed B
        a.q[0] = *(const uint4*)(Arow + k * 32 + hi * 16);
        a.q[1] = *(const uint4*)(Arow + k * 32 + hi * 16 + 8);
        c = __builtin_amdgcn_wmma_f32_16x16x32_f16(false, a.v, false, b[k].v,
                                                   (short)0, c, false, false);
      }
      // prefetch next block's A rows (global_prefetch_b8; speculative, OOB-safe)
      __builtin_prefetch(Arow + (size_t)64 * K, 0, 1);

      float* Cp = C + (size_t)(m0 + hi * 8) * DF + n0 + lo;
      if (m0 + 16 <= M) {
#pragma unroll
        for (int r = 0; r < 8; ++r) Cp[(size_t)r * DF] = c[r];
      } else {
#pragma unroll
        for (int r = 0; r < 8; ++r)
          if (m0 + hi * 8 + r < M) Cp[(size_t)r * DF] = c[r];
      }
    }
  }
}

// ---------- edge scatter: agg[dst] += h[src] * rsqrt(deg[src]*deg[dst]) ----------
__global__ void k_scatter(const int* __restrict__ src, const int* __restrict__ dst,
                          const float* __restrict__ deg, const float* __restrict__ h,
                          float* __restrict__ agg, int e) {
  int t = blockIdx.x * blockDim.x + threadIdx.x;
  int eid = t >> 4;
  if (eid >= e) return;
  int q = (t & 15) * 4;
  int s = src[eid], d = dst[eid];
  float coef = rsqrtf(deg[s] * deg[d]);
  float4 hv = *(const float4*)(h + (size_t)s * DF + q);
  float* ap = agg + (size_t)d * DF + q;
  atomicAdd(ap + 0, hv.x * coef);
  atomicAdd(ap + 1, hv.y * coef);
  atomicAdd(ap + 2, hv.z * coef);
  atomicAdd(ap + 3, hv.w * coef);
}

// combined = agg + h/deg + bias ; also accumulate GraphNorm stats
__global__ void k_combine_stats(float* __restrict__ agg, const float* __restrict__ hg,
                                const float* __restrict__ deg, const float* __restrict__ bias,
                                float* __restrict__ sum, float* __restrict__ sumsq, int n) {
  __shared__ float s1[TB], s2[TB];
  int t = threadIdx.x;
  int idx = blockIdx.x * TB + t;
  float v = 0.f;
  if (idx < n) {
    int row = idx >> 6, col = idx & 63;
    v = agg[idx] + hg[idx] * (1.0f / deg[row]) + bias[col];
    agg[idx] = v;
  }
  s1[t] = v; s2[t] = v * v;
  __syncthreads();
  if (t < 128) { s1[t] += s1[t + 128]; s2[t] += s2[t + 128]; }
  __syncthreads();
  if (t < 64) {
    atomicAdd(&sum[t],   s1[t] + s1[t + 64]);
    atomicAdd(&sumsq[t], s2[t] + s2[t + 64]);
  }
}

__global__ void k_norm_relu_f16(const float* __restrict__ comb, const float* __restrict__ mul,
                                const float* __restrict__ add, _Float16* __restrict__ out, int n) {
  int idx = blockIdx.x * blockDim.x + threadIdx.x;
  if (idx < n) {
    int col = idx & 63;
    float v = mul[col] * comb[idx] + add[col];
    out[idx] = (_Float16)fmaxf(v, 0.0f);
  }
}

__global__ void k_norm_relu_f32(const float* __restrict__ comb, const float* __restrict__ mul,
                                const float* __restrict__ add, float* __restrict__ out, int n) {
  int idx = blockIdx.x * blockDim.x + threadIdx.x;
  if (idx < n) {
    int col = idx & 63;
    float v = mul[col] * comb[idx] + add[col];
    out[idx] = fmaxf(v, 0.0f);
  }
}

// A2[l, 0:64] = h1[pos1[l,0]], A2[l, 64:128] = h1[pos1[l,1]]  (4 halves per thread)
__global__ void k_build_a2(const int* __restrict__ pos1, const _Float16* __restrict__ h1,
                           _Float16* __restrict__ a2, int n /* L*32 */) {
  int t = blockIdx.x * blockDim.x + threadIdx.x;
  if (t >= n) return;
  int l = t >> 5;
  int c0 = (t & 31) * 4;
  int srow = pos1[l * 2 + (c0 >= 64 ? 1 : 0)];
  *(uint2*)(a2 + (size_t)l * 128 + c0) =
      *(const uint2*)(h1 + (size_t)srow * DF + (c0 & 63));
}

// out[p] = dot(h2[pos2[p]], pW) + pb
__global__ void k_pred(const int* __restrict__ pos2, const float* __restrict__ h2,
                       const float* __restrict__ pW, const float* __restrict__ pb,
                       float* __restrict__ out, int P) {
  int p = blockIdx.x * blockDim.x + threadIdx.x;
  if (p >= P) return;
  const float* hr = h2 + (size_t)pos2[p] * DF;
  float s = 0.f;
#pragma unroll
  for (int q = 0; q < 16; ++q) {
    float4 a = *(const float4*)(hr + q * 4);
    float4 w = *(const float4*)(pW + q * 4);
    s += a.x * w.x + a.y * w.y + a.z * w.z + a.w * w.w;
  }
  out[p] = s + pb[0];
}

// ---------- launch ----------
extern "C" void kernel_launch(void* const* d_in, const int* in_sizes, int n_in,
                              void* d_out, int out_size, void* d_ws, size_t ws_size,
                              hipStream_t stream) {
  const int*   x    = (const int*)d_in[0];
  const int*   e1   = (const int*)d_in[1];
  const int*   e2   = (const int*)d_in[2];
  const int*   pos1 = (const int*)d_in[3];
  const int*   pos2 = (const int*)d_in[4];
  const float* embW = (const float*)d_in[5];
  const float* n0a = (const float*)d_in[6],  *n0g = (const float*)d_in[7],  *n0b = (const float*)d_in[8];
  const float* W1  = (const float*)d_in[9],  *b1  = (const float*)d_in[10];
  const float* n1a = (const float*)d_in[11], *n1g = (const float*)d_in[12], *n1b = (const float*)d_in[13];
  const float* W2  = (const float*)d_in[14], *b2  = (const float*)d_in[15];
  const float* n2a = (const float*)d_in[16], *n2g = (const float*)d_in[17], *n2b = (const float*)d_in[18];
  const float* pW  = (const float*)d_in[19], *pb  = (const float*)d_in[20];
  float* out = (float*)d_out;

  const int N  = in_sizes[0];
  const int E1 = in_sizes[1] / 2;
  const int E2 = in_sizes[2] / 2;
  const int L  = in_sizes[3] / 2;
  const int P  = in_sizes[4];
  const size_t maxNL = (size_t)(N > L ? N : L);

  // carve workspace
  char* w = (char*)d_ws;
  size_t off = 0;
  auto carve = [&](size_t bytes) -> char* {
    char* p = w + off;
    off += (bytes + 255) & ~(size_t)255;
    return p;
  };
  _Float16* hf16 = (_Float16*)carve((size_t)N * DF * 2);     // h0 f16, later h1n f16
  float*    hg   = (float*)   carve(maxNL * DF * 4);         // gemm out, later h2n
  float*    agg  = (float*)   carve(maxNL * DF * 4);         // scatter target / combined
  _Float16* a2   = (_Float16*)carve((size_t)L * 128 * 2);    // stage-2 A matrix
  float*    deg  = (float*)   carve(maxNL * 4);
  _Float16* Bp1  = (_Float16*)carve(2 * 2048 * 2);           // packed W1 frags (K=64)
  _Float16* Bp2  = (_Float16*)carve(4 * 2048 * 2);           // packed W2 frags (K=128)
  float*    st   = (float*)   carve(3 * 4 * 64 * 4);         // per stage: sum,sumsq,mul,add
  float *sum0 = st,       *sq0 = st + 64,  *mul0 = st + 128, *add0 = st + 192;
  float *sum1 = st + 256, *sq1 = st + 320, *mul1 = st + 384, *add1 = st + 448;
  float *sum2 = st + 512, *sq2 = st + 576, *mul2 = st + 640, *add2 = st + 704;

  auto cdiv = [](long long a, long long b) -> unsigned { return (unsigned)((a + b - 1) / b); };

  hipMemsetAsync(st, 0, 3 * 4 * 64 * 4, stream);
  k_pack_B<<<cdiv(2 * 2048, TB), TB, 0, stream>>>(W1, Bp1, 64);
  k_pack_B<<<cdiv(4 * 2048, TB), TB, 0, stream>>>(W2, Bp2, 128);

  // stage 0: embedding + GraphNorm -> f16
  const int nN = N * DF;
  k_stats_emb<<<cdiv(nN, TB), TB, 0, stream>>>(x, embW, sum0, sq0, nN);
  k_finalize<<<1, 64, 0, stream>>>(sum0, sq0, n0a, n0g, n0b, mul0, add0, 1.0f / (float)N);
  k_norm0<<<cdiv(nN, TB), TB, 0, stream>>>(x, embW, mul0, add0, hf16, nN);

  // stage 1: GCNConv(64->64) + GraphNorm + ReLU -> f16
  k_gemm_wmma<64><<<cdiv(N, 64), 128, 0, stream>>>(hf16, Bp1, hg, N);
  k_deg_init<<<cdiv(N, TB), TB, 0, stream>>>(deg, N);
  k_deg_count<<<cdiv(E1, TB), TB, 0, stream>>>(e1 + E1, deg, E1);
  hipMemsetAsync(agg, 0, (size_t)N * DF * 4, stream);
  k_scatter<<<cdiv((long long)E1 * 16, TB), TB, 0, stream>>>(e1, e1 + E1, deg, hg, agg, E1);
  k_combine_stats<<<cdiv(nN, TB), TB, 0, stream>>>(agg, hg, deg, b1, sum1, sq1, nN);
  k_finalize<<<1, 64, 0, stream>>>(sum1, sq1, n1a, n1g, n1b, mul1, add1, 1.0f / (float)N);
  k_norm_relu_f16<<<cdiv(nN, TB), TB, 0, stream>>>(agg, mul1, add1, hf16, nN);

  // pair gather -> [L,128] f16
  k_build_a2<<<cdiv((long long)L * 32, TB), TB, 0, stream>>>(pos1, hf16, a2, L * 32);

  // stage 2: GCNConv(128->64) + GraphNorm + ReLU -> f32
  const int nL = L * DF;
  k_gemm_wmma<128><<<cdiv(L, 64), 128, 0, stream>>>(a2, Bp2, hg, L);
  k_deg_init<<<cdiv(L, TB), TB, 0, stream>>>(deg, L);
  k_deg_count<<<cdiv(E2, TB), TB, 0, stream>>>(e2 + E2, deg, E2);
  hipMemsetAsync(agg, 0, (size_t)L * DF * 4, stream);
  k_scatter<<<cdiv((long long)E2 * 16, TB), TB, 0, stream>>>(e2, e2 + E2, deg, hg, agg, E2);
  k_combine_stats<<<cdiv(nL, TB), TB, 0, stream>>>(agg, hg, deg, b2, sum2, sq2, nL);
  k_finalize<<<1, 64, 0, stream>>>(sum2, sq2, n2a, n2g, n2b, mul2, add2, 1.0f / (float)L);
  k_norm_relu_f32<<<cdiv(nL, TB), TB, 0, stream>>>(agg, mul2, add2, hg, nL);

  // prediction head
  k_pred<<<cdiv(P, TB), TB, 0, stream>>>(pos2, hg, pW, pb, out, P);
}
